// PathTeamScorer_69741678952867
// MI455X (gfx1250) — matrix operand order repaired and checked by reading, not compiled, not run on hardware
//
#include <hip/hip_runtime.h>
#include <hip/hip_fp16.h>

typedef __attribute__((ext_vector_type(16))) _Float16 v16h;
typedef __attribute__((ext_vector_type(8)))  _Float16 v8h;
typedef __attribute__((ext_vector_type(8)))  float    v8f;

#define F_DIM   4096
#define T_DIM   9
#define H_DIM   32
#define KCHUNKS (F_DIM / 32)        // 128 chunks of K=32
#define CHUNK   (32 * T_DIM)        // 288 contiguous floats per chunk in [F,T] layout
#define ROWP    40                  // padded LDS row stride in halves (80 B) -> conflict-free b128
#define WAVES_PER_BLOCK 8

// ---------------------------------------------------------------------------
// Prep: fold theta1 rotation into W1 and pack as f16 WMMA B-fragments.
// Fragment layout: frag[((kc*2 + tile)*32 + lane)*16 + e], e = 0..15
// B matrix (K=32 x N=16, 16-bit): lanes 0-15 hold K={0..7,16..23} (col N=lane),
// lanes 16-31 hold K={8..15,24..31} (col N=lane-16). h = N + 16*tile, f = kc*32 + K.
// ---------------------------------------------------------------------------
__global__ void prep_w1_kernel(const float* __restrict__ W1,
                               const float* __restrict__ theta1,
                               _Float16* __restrict__ frag) {
    int idx  = blockIdx.x * blockDim.x + threadIdx.x;   // 128*2*32*16 = 131072
    int e    = idx & 15;
    int lane = (idx >> 4) & 31;
    int tile = (idx >> 9) & 1;
    int kc   = idx >> 10;
    int n    = lane & 15;
    int k    = ((e < 8) ? e : (e + 8)) + ((lane < 16) ? 0 : 8);
    int h    = n + (tile << 4);
    int f    = kc * 32 + k;
    float w;
    if (f < 4) {
        // rotation fold: W1eff[2p]   =  c*W1[2p] + s*W1[2p+1]
        //                W1eff[2p+1] = -s*W1[2p] + c*W1[2p+1]
        int   p  = f >> 1;
        float c  = cosf(theta1[p]);
        float s  = sinf(theta1[p]);
        float wa = W1[(2 * p) * H_DIM + h];
        float wb = W1[(2 * p + 1) * H_DIM + h];
        w = (f & 1) ? (c * wb - s * wa) : (c * wa + s * wb);
    } else {
        w = W1[f * H_DIM + h];
    }
    frag[idx] = (_Float16)w;
}

// ---------------------------------------------------------------------------
// Main: one wave per batch b. 128 iterations of K=32, two 16x16x32 f16 WMMAs.
// ---------------------------------------------------------------------------
__global__ void __launch_bounds__(256)
path_team_scorer_kernel(const float* __restrict__ tp,       // [B, F, T]
                        const float* __restrict__ b1,       // [32]
                        const float* __restrict__ gamma,    // [32]
                        const float* __restrict__ beta,     // [32]
                        const float* __restrict__ theta2,   // [2]
                        const float* __restrict__ W2,       // [32]
                        const _Float16* __restrict__ frag,  // packed W1eff fragments
                        float* __restrict__ out,            // [B, T]
                        int Btot) {
    __shared__ alignas(16) _Float16 smem[WAVES_PER_BLOCK][16 * ROWP];

    const int lane = threadIdx.x & 31;
    const int wid  = threadIdx.x >> 5;
    const int b    = blockIdx.x * WAVES_PER_BLOCK + wid;
    if (b >= Btot) return;                       // wave-uniform exit

    _Float16* tw = &smem[wid][0];

    // zero the tile once (rows t=9..15 stay zero; rows 0..8 rewritten each iter)
    for (int i = lane; i < 16 * ROWP; i += 32) tw[i] = (_Float16)0.0f;

    const float* ap = tp + (size_t)b * (F_DIM * T_DIM);

    // A-fragment gather addresses (16-bit A 16x32 layout):
    // lane L: row m = L%16; lanes<16 hold K={0..7,16..23}, lanes>=16 K={8..15,24..31}
    const int m   = lane & 15;
    const int klo = (lane < 16) ? 0 : 8;
    const _Float16* arow = tw + m * ROWP + klo;

    v8f acc0 = {};   // h = 0..15
    v8f acc1 = {};   // h = 16..31

    for (int kc = 0; kc < KCHUNKS; ++kc) {
        const float* cp = ap + kc * CHUNK;

        // coalesced load of the contiguous [32 x 9] f32 chunk (9 x 128B lines)
        float a[T_DIM];
#pragma unroll
        for (int i = 0; i < T_DIM; ++i) a[i] = cp[lane + 32 * i];

        if (kc + 1 < KCHUNKS)
            __builtin_prefetch(cp + CHUNK + lane * T_DIM, 0, 3);  // WGP-scope: next chunk -> all levels

        asm volatile("" ::: "memory");           // order: prev ds_loads before these stores
#pragma unroll
        for (int i = 0; i < T_DIM; ++i) {        // transpose-scatter into [t][k] f16 tile
            int flat = lane + 32 * i;
            int k = flat / T_DIM;
            int t = flat - T_DIM * k;
            tw[t * ROWP + k] = (_Float16)a[i];
        }
        asm volatile("" ::: "memory");           // order: stores before ds_loads (LDS in-order per wave)

        // A fragment: two ds_load_b128 per lane
        v8h alo = *(const v8h*)(arow);           // K = klo .. klo+7
        v8h ahi = *(const v8h*)(arow + 16);      // K = klo+16 .. klo+23
        v16h A  = __builtin_shufflevector(alo, ahi, 0, 1, 2, 3, 4, 5, 6, 7,
                                                    8, 9, 10, 11, 12, 13, 14, 15);

        // B fragments: pre-packed f16, L2-resident
        const _Float16* bp = frag + ((size_t)(kc * 2) * 32 + lane) * 16;
        v16h B0 = *(const v16h*)(bp);            // h = 0..15
        v16h B1 = *(const v16h*)(bp + 32 * 16);  // h = 16..31

        acc0 = __builtin_amdgcn_wmma_f32_16x16x32_f16(false, A, false, B0,
                                                      (short)0, acc0, false, false);
        acc1 = __builtin_amdgcn_wmma_f32_16x16x32_f16(false, A, false, B1,
                                                      (short)0, acc1, false, false);
    }

    // ---- epilogue: bias + ModularPhaseNorm + (theta2-rotated W2) dot -------
    // C/D layout: lane L holds col n=L%16; VGPR v holds row t = v + 8*(L/16)
    const int n = lane & 15;
    float rb1[2], rg[2], rbt[2], rw2[2];
#pragma unroll
    for (int j = 0; j < 2; ++j) {
        int h = n + 16 * j;
        rb1[j] = b1[h];
        rg[j]  = gamma[h];
        rbt[j] = beta[h];
        float w;
        if (h < 4) {   // fold theta2: W2eff[2p] = c*W2[2p]+s*W2[2p+1]; [2p+1] = -s*W2[2p]+c*W2[2p+1]
            int   p  = h >> 1;
            float c  = cosf(theta2[p]);
            float s  = sinf(theta2[p]);
            float wa = W2[2 * p], wb = W2[2 * p + 1];
            w = (h & 1) ? (c * wb - s * wa) : (c * wa + s * wb);
        } else {
            w = W2[h];
        }
        rw2[j] = w;
    }

    float sc[8];
#pragma unroll
    for (int v = 0; v < 8; ++v) {
        float x0 = acc0[v] + rb1[0];
        x0 -= 7.0f * floorf(x0 * (1.0f / 7.0f) + 0.5f);
        float y = (rg[0] * x0 + rbt[0]) * rw2[0];
        float x1 = acc1[v] + rb1[1];
        x1 -= 7.0f * floorf(x1 * (1.0f / 7.0f) + 0.5f);
        y += (rg[1] * x1 + rbt[1]) * rw2[1];
        // reduce over the 16 N-lanes within each half-wave (b2 cancels in softmax)
        y += __shfl_xor(y, 1);
        y += __shfl_xor(y, 2);
        y += __shfl_xor(y, 4);
        y += __shfl_xor(y, 8);
        sc[v] = y;   // half0: t = v ; half1: t = v + 8
    }
    float s8 = __shfl(sc[0], 16);   // t = 8 lives in the upper half-wave

    if (lane == 0) {
        float s[T_DIM];
#pragma unroll
        for (int i = 0; i < 8; ++i) s[i] = sc[i];
        s[8] = s8;
        float mx = s[0];
#pragma unroll
        for (int i = 1; i < T_DIM; ++i) mx = fmaxf(mx, s[i]);
        float e[T_DIM], sum = 0.0f;
#pragma unroll
        for (int i = 0; i < T_DIM; ++i) { e[i] = __expf(s[i] - mx); sum += e[i]; }
        float inv = 1.0f / sum;
#pragma unroll
        for (int i = 0; i < T_DIM; ++i) out[(size_t)b * T_DIM + i] = e[i] * inv;
    }
}

extern "C" void kernel_launch(void* const* d_in, const int* in_sizes, int n_in,
                              void* d_out, int out_size, void* d_ws, size_t ws_size,
                              hipStream_t stream) {
    const float* tp     = (const float*)d_in[0];
    const float* theta1 = (const float*)d_in[1];
    const float* W1     = (const float*)d_in[2];
    const float* b1     = (const float*)d_in[3];
    const float* gamma  = (const float*)d_in[4];
    const float* beta   = (const float*)d_in[5];
    const float* theta2 = (const float*)d_in[6];
    const float* W2     = (const float*)d_in[7];
    float*       out    = (float*)d_out;
    _Float16*    frag   = (_Float16*)d_ws;      // 128*2*32*16 f16 = 256 KB

    int Btot = in_sizes[0] / (F_DIM * T_DIM);   // 8192

    // 131072 fragment elements, one thread each
    prep_w1_kernel<<<512, 256, 0, stream>>>(W1, theta1, frag);

    int blocks = (Btot + WAVES_PER_BLOCK - 1) / WAVES_PER_BLOCK;
    path_team_scorer_kernel<<<blocks, 256, 0, stream>>>(
        tp, b1, gamma, beta, theta2, W2, frag, out, Btot);
}
